// BahdanauAttention_8315056685070
// MI455X (gfx1250) — matrix operand order repaired and checked by reading, not compiled, not run on hardware
//
#include <hip/hip_runtime.h>
#include <hip/hip_bf16.h>
#include <math.h>

// ---------------------------------------------------------------------------
// Bahdanau-style attention for MI455X (gfx1250, wave32, WMMA bf16).
// Shapes: L=4, T=2048, C=32, QL=KL=512, H=256.
// ---------------------------------------------------------------------------

typedef __attribute__((ext_vector_type(16))) __bf16 v16bf;
typedef __attribute__((ext_vector_type(8)))  __bf16 v8bf;
typedef __attribute__((ext_vector_type(8)))  float  v8f;

#define L_   4
#define T_   2048
#define C_   32
#define KL_  512
#define QL_  512
#define H_   256
#define RPL  (T_ * C_)        // rows per layer in the K-projection GEMM = 65536
#define LT_  (L_ * T_)        // 8192
#define NCHUNK 32             // context partial-sum chunks (8192/256 rows each)

#define BROW  (KL_ + 8)                   // padded LDS row stride (bf16): bank 4n mod 64
#define LDS_BYTES (H_ * BROW * 2)         // 256 * 520 * 2 = 266,240 B (< 320 KB/WGP)

// ---------------------------------------------------------------------------
// Kernel 1: convert Kw [L,H,KL] f32 -> bf16 (L2-resident B operand source)
// ---------------------------------------------------------------------------
__global__ void cvt_kw_bf16(const float* __restrict__ kw, __bf16* __restrict__ out, int n) {
    int i = blockIdx.x * blockDim.x + threadIdx.x;
    if (i < n) out[i] = (__bf16)kw[i];
}

// ---------------------------------------------------------------------------
// Kernel 2: q projection  qh[l,c,h] = sum_q queries[l,0,c,q]*Qw[l,h,q] + Qb[l,h]
// Tiny (32768 outputs x 512 MACs) — scalar is fine.
// ---------------------------------------------------------------------------
__global__ void qproj_kernel(const float* __restrict__ queries,
                             const float* __restrict__ Qw,
                             const float* __restrict__ Qb,
                             float* __restrict__ qh) {
    int idx = blockIdx.x * blockDim.x + threadIdx.x;   // [l][c][h]
    if (idx >= L_ * C_ * H_) return;
    int h = idx & (H_ - 1);
    int c = (idx >> 8) & (C_ - 1);
    int l = idx >> 13;
    const float* qr = queries + ((size_t)l * C_ + c) * QL_;
    const float* wr = Qw + ((size_t)l * H_ + h) * QL_;
    float s = 0.f;
    for (int i = 0; i < QL_; ++i) s += qr[i] * wr[i];
    qh[idx] = s + Qb[l * H_ + h];
}

// ---------------------------------------------------------------------------
// Kernel 3 (main): per-layer GEMM  K = keys[l] (65536x512) x Kw[l]^T (512x256)
// via v_wmma_f32_16x16x32_bf16, fused with  h = tanh(q + k + Kb)  and
// scores = h . Vw + Vb  (256-long dot reduced with lane shuffles).
//
// B operand (Kw[l], bf16) is staged ONCE per block into LDS (260 KB of the
// 320 KB WGP LDS) with a bank-conflict-free padded stride; the hot loop reads
// it with ds_load_b128. K-loop unrolled x2 to widen the independent-work
// window (only 2 waves/SIMD at this LDS footprint).
//
// One wave: M=16 rows x N=256 (16 WMMA acc tiles), K-loop step 32.
// Block = 256 threads = 8 waves = 128 rows. Grid: (512, L).
// ---------------------------------------------------------------------------
__global__ __launch_bounds__(256, 1)
void kproj_scores_kernel(const float* __restrict__ keys,     // [L][RPL][KL]
                         const __bf16* __restrict__ kw_bf,   // [L][H][KL]
                         const float* __restrict__ Kb,       // [L][H]
                         const float* __restrict__ qh,       // [L][C][H]
                         const float* __restrict__ Vw,       // [L][H]
                         const float* __restrict__ Vb,       // [L]
                         float* __restrict__ scores)         // [L][RPL]
{
    extern __shared__ __align__(16) char smem_raw[];
    __bf16* Bs = (__bf16*)smem_raw;                          // [H_][BROW]

    const int l    = blockIdx.y;
    const int wave = threadIdx.x >> 5;
    const int lane = threadIdx.x & 31;
    const int m0   = (blockIdx.x * 8 + wave) * 16;           // row-tile base (per layer)

    // ---- Stage Kw[l] (bf16) into LDS: 256 rows x 512 bf16, padded stride ----
    {
        const __bf16* src = kw_bf + (size_t)l * H_ * KL_;
        // 16-byte chunks: 64 chunks per row, 256 rows -> 16384 chunks / 256 thr
        for (int i = threadIdx.x; i < H_ * (KL_ / 8); i += 256) {
            const int row = i >> 6;          // / 64
            const int ch  = i & 63;
            *(float4*)(Bs + (size_t)row * BROW + ch * 8) =
                *(const float4*)(src + (size_t)row * KL_ + ch * 8);
        }
        __syncthreads();
    }

    const float* A = keys + (size_t)l * RPL * KL_;
    const int lhalf = (lane < 16) ? 0 : 1;
    const int nlo   = lane & 15;
    const int mrow  = m0 + nlo;
    const float* arow = A + (size_t)mrow * KL_;
    // Per-lane LDS base for B fragments: column row nlo, half-dependent K shift (8).
    const __bf16* Blane = Bs + (size_t)nlo * BROW + lhalf * 8;

    v8f acc[16];
#pragma unroll
    for (int i = 0; i < 16; ++i) acc[i] = (v8f){0.f,0.f,0.f,0.f,0.f,0.f,0.f,0.f};

#pragma unroll 2
    for (int kk = 0; kk < KL_ / 32; ++kk) {
        const int kbase = kk * 32;
        // ---- A fragment: ISA 16-bit A 16x32 layout ----
        // lane<16 : K = kbase+{0..7, 16..23};  lane>=16 : K = kbase+{8..15, 24..31}
        const int ka = kbase + lhalf * 8;
        float4 a0 = *(const float4*)(arow + ka);
        float4 a1 = *(const float4*)(arow + ka + 4);
        float4 a2 = *(const float4*)(arow + ka + 16);
        float4 a3 = *(const float4*)(arow + ka + 20);
        v16bf av;
        av[0]=(__bf16)a0.x;  av[1]=(__bf16)a0.y;  av[2]=(__bf16)a0.z;  av[3]=(__bf16)a0.w;
        av[4]=(__bf16)a1.x;  av[5]=(__bf16)a1.y;  av[6]=(__bf16)a1.z;  av[7]=(__bf16)a1.w;
        av[8]=(__bf16)a2.x;  av[9]=(__bf16)a2.y;  av[10]=(__bf16)a2.z; av[11]=(__bf16)a2.w;
        av[12]=(__bf16)a3.x; av[13]=(__bf16)a3.y; av[14]=(__bf16)a3.z; av[15]=(__bf16)a3.w;

        // ---- B fragments from LDS: lane holds column n = nt*16 + nlo,
        //      K = kbase+{0..7,16..23} (+8 for the upper half-wave), matching A.
#pragma unroll
        for (int nt = 0; nt < 16; ++nt) {
            const __bf16* bp = Blane + (size_t)nt * 16 * BROW + kbase;
            v8bf blo = *(const v8bf*)(bp);           // 16B-aligned ds_load_b128
            v8bf bhi = *(const v8bf*)(bp + 16);
            v16bf bv = __builtin_shufflevector(blo, bhi,
                0,1,2,3,4,5,6,7,8,9,10,11,12,13,14,15);
            acc[nt] = __builtin_amdgcn_wmma_f32_16x16x32_bf16(
                false, av, false, bv, (short)0, acc[nt], false, false);
        }
    }

    // ---- Fused epilogue ----
    // C/D layout: lane p: n = nt*16 + (p&15); VGPR r: row m = r + (p<16?0:8).
    const float* kbl = Kb + (size_t)l * H_;
    const float* vwl = Vw + (size_t)l * H_;
    const float* qhl = qh + (size_t)l * C_ * H_;
    float part[8];
#pragma unroll
    for (int r = 0; r < 8; ++r) part[r] = 0.f;

#pragma unroll
    for (int nt = 0; nt < 16; ++nt) {
        const int n   = nt * 16 + nlo;
        const float kbv = kbl[n];
        const float vwv = vwl[n];
#pragma unroll
        for (int r = 0; r < 8; ++r) {
            const int row = m0 + r + lhalf * 8;       // row = t*C + c within layer
            const int c   = row & (C_ - 1);
            const float qv = qhl[(size_t)c * H_ + n];
            const float hv = tanhf(acc[nt][r] + kbv + qv);
            part[r] += hv * vwv;
        }
    }
    // Reduce over the 16 lanes of each half (lanes 0-15 hold rows 0-7, 16-31 rows 8-15).
#pragma unroll
    for (int off = 8; off >= 1; off >>= 1) {
#pragma unroll
        for (int r = 0; r < 8; ++r) part[r] += __shfl_xor(part[r], off, 32);
    }
    if (nlo == 0) {
        const float vbv = Vb[l];
#pragma unroll
        for (int r = 0; r < 8; ++r) {
            const int row = m0 + r + lhalf * 8;
            scores[(size_t)l * RPL + row] = part[r] + vbv;
        }
    }
}

// ---------------------------------------------------------------------------
// Kernel 4: global softmax over the flattened (L*T) axis per column c.
// scores viewed as [LT_=8192][C_=32]. One block per column.
// ---------------------------------------------------------------------------
__global__ void softmax_kernel(const float* __restrict__ scores,
                               float* __restrict__ wts) {   // wts = d_out + 16384
    const int c   = blockIdx.x;
    const int tid = threadIdx.x;                 // 256
    __shared__ float red[256];

    float m = -INFINITY;
    for (int R = tid; R < LT_; R += 256) m = fmaxf(m, scores[(size_t)R * C_ + c]);
    red[tid] = m; __syncthreads();
    for (int s = 128; s > 0; s >>= 1) { if (tid < s) red[tid] = fmaxf(red[tid], red[tid + s]); __syncthreads(); }
    m = red[0]; __syncthreads();

    float sum = 0.f;
    for (int R = tid; R < LT_; R += 256) sum += expf(scores[(size_t)R * C_ + c] - m);
    red[tid] = sum; __syncthreads();
    for (int s = 128; s > 0; s >>= 1) { if (tid < s) red[tid] += red[tid + s]; __syncthreads(); }
    const float inv = 1.f / red[0];

    for (int R = tid; R < LT_; R += 256)
        wts[(size_t)R * C_ + c] = expf(scores[(size_t)R * C_ + c] - m) * inv;
}

// ---------------------------------------------------------------------------
// Kernel 5: context partial sums (deterministic, no float atomics).
// context[c,k] = sum_R wts[R,c]*keys_flat[(R*C+c)*KL + k]; split R into chunks.
// Grid: (C_, NCHUNK), block 256 threads (each thread covers k and k+256).
// ---------------------------------------------------------------------------
__global__ void context_partial_kernel(const float* __restrict__ keys,
                                       const float* __restrict__ wts,
                                       float* __restrict__ part) { // [NCHUNK][C_][KL_]
    const int c     = blockIdx.x;
    const int chunk = blockIdx.y;
    const int k     = threadIdx.x;
    float a0 = 0.f, a1 = 0.f;
    const int r0 = chunk * (LT_ / NCHUNK);
    for (int i = 0; i < LT_ / NCHUNK; ++i) {
        const int R = r0 + i;
        const float w = wts[(size_t)R * C_ + c];
        const float* kr = keys + ((size_t)R * C_ + c) * KL_;
        a0 += w * kr[k];
        a1 += w * kr[k + 256];
    }
    float* p = part + ((size_t)chunk * C_ + c) * KL_;
    p[k]       = a0;
    p[k + 256] = a1;
}

__global__ void context_reduce_kernel(const float* __restrict__ part,
                                      float* __restrict__ ctx) {   // [C_*KL_]
    const int idx = blockIdx.x * blockDim.x + threadIdx.x;         // 16384
    if (idx >= C_ * KL_) return;
    float s = 0.f;
    for (int j = 0; j < NCHUNK; ++j) s += part[(size_t)j * C_ * KL_ + idx];
    ctx[idx] = s;
}

// ---------------------------------------------------------------------------
// Launcher. Inputs (setup_inputs order):
// 0 queries [L,1,C,QL], 1 keys [L,T,C,KL], 2 Qw [L,H,QL], 3 Qb [L,H],
// 4 Kw [L,H,KL], 5 Kb [L,H], 6 Vw [L,H], 7 Vb [L].
// d_out: context (C*KL=16384 floats) then weights (L*T*C=262144 floats).
// ---------------------------------------------------------------------------
extern "C" void kernel_launch(void* const* d_in, const int* in_sizes, int n_in,
                              void* d_out, int out_size, void* d_ws, size_t ws_size,
                              hipStream_t stream) {
    const float* queries = (const float*)d_in[0];
    const float* keys    = (const float*)d_in[1];
    const float* Qw      = (const float*)d_in[2];
    const float* Qb      = (const float*)d_in[3];
    const float* Kw      = (const float*)d_in[4];
    const float* Kb      = (const float*)d_in[5];
    const float* Vw      = (const float*)d_in[6];
    const float* Vb      = (const float*)d_in[7];

    float* ctx = (float*)d_out;                  // [C_*KL_] = 16384
    float* wts = (float*)d_out + C_ * KL_;       // [LT_*C_] = 262144

    // Workspace layout (bytes):
    //   kw_bf16 : 4*256*512*2   = 1,048,576
    //   qh      : 4*32*256*4    =   131,072
    //   scores  : 4*65536*4     = 1,048,576
    //   partials: 32*32*512*4   = 2,097,152   (total ~4.2 MB)
    char* ws = (char*)d_ws;
    __bf16* kw_bf  = (__bf16*)ws;
    float*  qh     = (float*)(ws + 1048576);
    float*  scores = (float*)(ws + 1179648);
    float*  part   = (float*)(ws + 2228224);

    // 1. Kw f32 -> bf16
    {
        int n = L_ * H_ * KL_;                   // 524288
        cvt_kw_bf16<<<(n + 255) / 256, 256, 0, stream>>>(Kw, kw_bf, n);
    }
    // 2. q projection (+Qb)
    {
        int n = L_ * C_ * H_;                    // 32768
        qproj_kernel<<<(n + 255) / 256, 256, 0, stream>>>(queries, Qw, Qb, qh);
    }
    // 3. Main fused WMMA GEMM + tanh + score dot (Kw staged in 260 KB of LDS)
    {
        dim3 grid(RPL / 128, L_);                // (512, 4)
        kproj_scores_kernel<<<grid, 256, LDS_BYTES, stream>>>(keys, kw_bf, Kb, qh, Vw, Vb, scores);
    }
    // 4. Global softmax per column -> weights (straight into d_out)
    softmax_kernel<<<C_, 256, 0, stream>>>(scores, wts);
    // 5-6. Context = weighted sum of keys (second keys pass), deterministic reduce
    {
        dim3 grid(C_, NCHUNK);
        context_partial_kernel<<<grid, 256, 0, stream>>>(keys, wts, part);
        context_reduce_kernel<<<(C_ * KL_ + 255) / 256, 256, 0, stream>>>(part, ctx);
    }
}
